// gqa_38053410242951
// MI455X (gfx1250) — matrix operand order repaired and checked by reading, not compiled, not run on hardware
//
#include <hip/hip_runtime.h>
#include <hip/hip_bf16.h>
#include <math.h>

// ---------------------------------------------------------------------------
// GQA forward on gfx1250: bf16 WMMA GEMMs (async-LDS staged A) + flash attn.
// B=2, S=1024, HIDDEN=2048, HEADS=32, GROUPS=8, HEAD_DIM=64.
// ---------------------------------------------------------------------------

typedef __bf16 bf16;
typedef __attribute__((ext_vector_type(16))) __bf16 v16bf;
typedef __attribute__((ext_vector_type(8)))  __bf16 v8bf;
typedef __attribute__((ext_vector_type(8)))  float  v8f;

#define SEQ     1024
#define HID     2048
#define KVDIM   512
#define NHEADS  32
#define NGROUPS 8
#define HD      64

static __device__ __forceinline__ v16bf cat8(v8bf lo, v8bf hi) {
  return __builtin_shufflevector(lo, hi, 0,1,2,3,4,5,6,7,8,9,10,11,12,13,14,15);
}

// ---- 16-lane row reductions via DPP row_xmask (pure VALU, no LDS) ---------
template <int M>
static __device__ __forceinline__ float dpp_xor(float x) {
  return __builtin_bit_cast(float,
      __builtin_amdgcn_update_dpp(0, __builtin_bit_cast(int, x),
                                  0x160 | M, 0xF, 0xF, true));
}
static __device__ __forceinline__ float rowmax16(float x) {
  x = fmaxf(x, dpp_xor<1>(x));
  x = fmaxf(x, dpp_xor<2>(x));
  x = fmaxf(x, dpp_xor<4>(x));
  x = fmaxf(x, dpp_xor<8>(x));
  return x;
}
static __device__ __forceinline__ float rowsum16(float x) {
  x += dpp_xor<1>(x);
  x += dpp_xor<2>(x);
  x += dpp_xor<4>(x);
  x += dpp_xor<8>(x);
  return x;
}

// ---------------------------------------------------------------------------
// fp32 -> bf16 conversion (memory bound, 4 elements/thread)
// ---------------------------------------------------------------------------
__global__ void f32_to_bf16_kernel(const float* __restrict__ src,
                                   bf16* __restrict__ dst, int n) {
  int i = (blockIdx.x * blockDim.x + threadIdx.x) * 4;
  if (i + 3 < n) {
    float4 f = *(const float4*)(src + i);
    dst[i + 0] = (bf16)f.x;
    dst[i + 1] = (bf16)f.y;
    dst[i + 2] = (bf16)f.z;
    dst[i + 3] = (bf16)f.w;
  }
}

// ---------------------------------------------------------------------------
// C[M,N] = A[M,K](bf16) * W[N,K]^T(bf16) + bias[N]
// STORE_MODE 0: bf16 row-major [M,N]
// STORE_MODE 1: bf16 transposed for V: [b, g, d, s]   (M = b*SEQ+s, n = g*64+d)
// STORE_MODE 2: f32 row-major [M,N]
// Block: 256 threads = 8 waves, block tile 64(M) x 256(N), wave tile 32x64.
// A tiles (2 m-groups x 32 rows x 64 K, shared by 4 waves each) are double-
// buffered in LDS via GLOBAL_LOAD_ASYNC_TO_LDS_B128; each stage feeds two
// unrolled 32-K WMMA sub-steps (16 WMMAs) per block barrier.
// ---------------------------------------------------------------------------
template <int STORE_MODE>
__global__ __launch_bounds__(256)
void gemm_bf16_wmma(const bf16* __restrict__ A, const bf16* __restrict__ W,
                    const float* __restrict__ bias, void* __restrict__ Cout,
                    int N, int K) {
  // atile[buf][mgroup][32 rows][64 K]  (2*2*4KB = 16KB)
  __shared__ __align__(16) bf16 atile[2][2][32 * 64];

  const int lane  = threadIdx.x & 31;
  const int wid   = threadIdx.x >> 5;
  const int col16 = lane & 15;
  const int half  = lane >> 4;
  const int mg    = wid >> 2;                  // m-group (0/1) of this wave
  const bool stager = ((wid & 3) == 0);        // one staging wave per m-group
  const int m0 = blockIdx.x * 64 + mg * 32;
  const int n0 = blockIdx.y * 256 + (wid & 3) * 64;

  // issue 8 x 16B async global->LDS copies covering 32 rows x 64 K of A
  auto stage = [&](int buf, int kk) {
    if (stager) {
#pragma unroll
      for (int i = 0; i < 8; ++i) {
        const int ci = lane + 32 * i;          // 256 chunks of 16B
        const int r  = ci >> 3;                // tile row
        const int ko = (ci & 7) * 8;           // K offset (elements)
        const bf16* src = A + (size_t)(m0 + r) * K + kk + ko;
        const unsigned dst =
            (unsigned)(uintptr_t)&atile[buf][mg][r * 64 + ko];
        asm volatile("global_load_async_to_lds_b128 %0, %1, off"
                     :: "v"(dst), "v"(src) : "memory");
      }
    }
  };

  v8f acc[2][4] = {};

  stage(0, 0);
  if (stager) asm volatile("s_wait_asynccnt 0" ::: "memory");
  __syncthreads();

  for (int kk = 0; kk < K; kk += 64) {
    const int buf = (kk >> 6) & 1;
    if (kk + 64 < K) stage(buf ^ 1, kk + 64);  // DMA next stage during compute

    const bf16* at = &atile[buf][mg][0];
#pragma unroll
    for (int ks = 0; ks < 64; ks += 32) {
      // A fragments (16x32 each) from LDS, layout identical to global case
      v16bf af0 = cat8(*(const v8bf*)(at + col16 * 64 + ks + 16 * half),
                       *(const v8bf*)(at + col16 * 64 + ks + 16 * half + 8));
      v16bf af1 = cat8(*(const v8bf*)(at + (16 + col16) * 64 + ks + 16 * half),
                       *(const v8bf*)(at + (16 + col16) * 64 + ks + 16 * half + 8));
#pragma unroll
      for (int j = 0; j < 4; ++j) {
        // B fragment (32x16): B[k][n] = W[n][k]; lane = column n, K contiguous
        const bf16* wrow =
            W + (size_t)(n0 + j * 16 + col16) * K + kk + ks + 16 * half;
        v16bf bfrag = *(const v16bf*)wrow;
        if (kk + 192 < K && ks == 0)
          __builtin_prefetch(wrow + 128, 0, 1);  // global_prefetch_b8
        acc[0][j] = __builtin_amdgcn_wmma_f32_16x16x32_bf16(
            false, af0, false, bfrag, (short)0, acc[0][j], false, false);
        acc[1][j] = __builtin_amdgcn_wmma_f32_16x16x32_bf16(
            false, af1, false, bfrag, (short)0, acc[1][j], false, false);
      }
    }

    if (stager) asm volatile("s_wait_asynccnt 0" ::: "memory");
    __syncthreads();
  }

#pragma unroll
  for (int j = 0; j < 4; ++j) {
    const int n = n0 + j * 16 + col16;
    const float bv = bias[n];
#pragma unroll
    for (int mi = 0; mi < 2; ++mi) {
#pragma unroll
      for (int v = 0; v < 8; ++v) {
        const int m = m0 + mi * 16 + v + 8 * half;  // C layout: lane=col, VGPR v=row
        const float val = acc[mi][j][v] + bv;
        if constexpr (STORE_MODE == 0) {
          ((bf16*)Cout)[(size_t)m * N + n] = (bf16)val;
        } else if constexpr (STORE_MODE == 1) {
          const int b = m / SEQ, s = m % SEQ;
          const int g = n / HD, d = n % HD;
          ((bf16*)Cout)[((size_t)(b * NGROUPS + g) * HD + d) * SEQ + s] = (bf16)val;
        } else {
          ((float*)Cout)[(size_t)m * N + n] = val;
        }
      }
    }
  }
}

// ---------------------------------------------------------------------------
// Causal flash attention. Grid: (SEQ/64, NHEADS, B); 128 threads = 4 waves,
// each wave owns 16 query rows; key step 32; online softmax in f32 with DPP
// row reductions.
// Qb: [B,S,HID] bf16 ; Kb: [B,S,KVDIM] bf16 ; Vt: [B,G,HD,S] bf16
// Ao: [B,S,HID] bf16
// ---------------------------------------------------------------------------
__global__ __launch_bounds__(128)
void gqa_flash_attn(const bf16* __restrict__ Qb, const bf16* __restrict__ Kb,
                    const bf16* __restrict__ Vt, bf16* __restrict__ Ao) {
  __shared__ __align__(32) bf16 lds_p[4][16 * 32];   // per-wave P tile staging

  const int lane  = threadIdx.x & 31;
  const int wid   = threadIdx.x >> 5;
  const int col16 = lane & 15;
  const int half  = lane >> 4;
  const int m0 = blockIdx.x * 64 + wid * 16;
  const int h  = blockIdx.y;
  const int b  = blockIdx.z;
  const int g  = h >> 2;
  const float scale = 0.125f;  // 64^-0.5

  // Q fragments: 16x64 = two 16x32 A-fragments
  const bf16* qrow = Qb + (size_t)(b * SEQ + m0 + col16) * HID + h * HD;
  const v16bf qf0 = cat8(*(const v8bf*)(qrow + 16 * half),
                         *(const v8bf*)(qrow + 16 * half + 8));
  const v16bf qf1 = cat8(*(const v8bf*)(qrow + 32 + 16 * half),
                         *(const v8bf*)(qrow + 40 + 16 * half));

  v8f o[4] = {};
  float mrow[8], lrow[8];
#pragma unroll
  for (int v = 0; v < 8; ++v) { mrow[v] = -__builtin_inff(); lrow[v] = 0.f; }

  const bf16* kbase = Kb + (size_t)b * SEQ * KVDIM + g * HD;
  const bf16* vbase = Vt + (size_t)(b * NGROUPS + g) * HD * SEQ;
  bf16* lp = lds_p[wid];
  const int jend = m0 + 16;

  for (int j0 = 0; j0 < jend; j0 += 32) {
    // ---- scores: two 16x16 tiles (keys j0..+15, j0+16..+31) ----
    v8f s0 = {}, s1 = {};
    {
      const bf16* k0 = kbase + (size_t)(j0 + col16) * KVDIM + 16 * half;
      s0 = __builtin_amdgcn_wmma_f32_16x16x32_bf16(false, qf0, false,
            *(const v16bf*)k0,        (short)0, s0, false, false);
      s0 = __builtin_amdgcn_wmma_f32_16x16x32_bf16(false, qf1, false,
            *(const v16bf*)(k0 + 32), (short)0, s0, false, false);
      const bf16* k1 = kbase + (size_t)(j0 + 16 + col16) * KVDIM + 16 * half;
      s1 = __builtin_amdgcn_wmma_f32_16x16x32_bf16(false, qf0, false,
            *(const v16bf*)k1,        (short)0, s1, false, false);
      s1 = __builtin_amdgcn_wmma_f32_16x16x32_bf16(false, qf1, false,
            *(const v16bf*)(k1 + 32), (short)0, s1, false, false);
    }

    const bool need_mask = (j0 + 32 >= jend);  // only final tile crosses diag
    float p0[8], p1[8];
#pragma unroll
    for (int v = 0; v < 8; ++v) {
      float a = s0[v] * scale, c = s1[v] * scale;
      const int row = m0 + v + 8 * half;
      if (need_mask) {
        if (j0 + col16 > row)      a = -__builtin_inff();
        if (j0 + 16 + col16 > row) c = -__builtin_inff();
      }
      // row max / sum across the 16 lanes holding this row's columns (DPP)
      const float rm = rowmax16(fmaxf(a, c));
      const float mn = fmaxf(mrow[v], rm);
      const float e0 = __expf(a - mn), e1 = __expf(c - mn);
      const float cr = __expf(mrow[v] - mn);
      const float rs = rowsum16(e0 + e1);
      lrow[v] = lrow[v] * cr + rs;
      mrow[v] = mn;
      p0[v] = e0; p1[v] = e1;
#pragma unroll
      for (int t = 0; t < 4; ++t) o[t][v] *= cr;
    }

    // ---- P (C-layout) -> A-fragment layout via wave-private LDS ----
#pragma unroll
    for (int v = 0; v < 8; ++v) {
      const int r = v + 8 * half;
      lp[r * 32 + col16]      = (bf16)p0[v];
      lp[r * 32 + 16 + col16] = (bf16)p1[v];
    }
    __builtin_amdgcn_wave_barrier();
    asm volatile("s_wait_dscnt 0" ::: "memory");  // in-wave LDS RAW fence
    const v16bf pf = cat8(*(const v8bf*)(lp + col16 * 32 + 16 * half),
                          *(const v8bf*)(lp + col16 * 32 + 16 * half + 8));
    __builtin_amdgcn_wave_barrier();

    // ---- PV: out[16x64] += P[16x32] * V[32x64] ----
#pragma unroll
    for (int t = 0; t < 4; ++t) {
      const bf16* vp = vbase + (size_t)(t * 16 + col16) * SEQ + j0 + 16 * half;
      o[t] = __builtin_amdgcn_wmma_f32_16x16x32_bf16(
          false, pf, false, *(const v16bf*)vp, (short)0, o[t], false, false);
    }
  }

  // ---- normalize and store ----
#pragma unroll
  for (int v = 0; v < 8; ++v) {
    const float inv = 1.0f / lrow[v];
    const size_t row = (size_t)(b * SEQ + m0 + v + 8 * half);
#pragma unroll
    for (int t = 0; t < 4; ++t) {
      Ao[row * HID + h * HD + t * 16 + col16] = (bf16)(o[t][v] * inv);
    }
  }
}

// ---------------------------------------------------------------------------
// launch
// ---------------------------------------------------------------------------
extern "C" void kernel_launch(void* const* d_in, const int* in_sizes, int n_in,
                              void* d_out, int out_size, void* d_ws, size_t ws_size,
                              hipStream_t stream) {
  const float* x  = (const float*)d_in[0];
  // d_in[1] = mask (causality is hardcoded)
  const float* Wq = (const float*)d_in[2];
  const float* bq = (const float*)d_in[3];
  const float* Wk = (const float*)d_in[4];
  const float* bk = (const float*)d_in[5];
  const float* Wv = (const float*)d_in[6];
  const float* bv = (const float*)d_in[7];
  const float* Wo = (const float*)d_in[8];
  const float* bo = (const float*)d_in[9];
  float* out = (float*)d_out;

  const int M = 2 * SEQ;  // 2048 tokens
  char* w = (char*)d_ws;
  bf16* Xb  = (bf16*)w; w += (size_t)M * HID * 2;
  bf16* Wqb = (bf16*)w; w += (size_t)HID * HID * 2;
  bf16* Wkb = (bf16*)w; w += (size_t)KVDIM * HID * 2;
  bf16* Wvb = (bf16*)w; w += (size_t)KVDIM * HID * 2;
  bf16* Wob = (bf16*)w; w += (size_t)HID * HID * 2;
  bf16* Qb  = (bf16*)w; w += (size_t)M * HID * 2;
  bf16* Kb  = (bf16*)w; w += (size_t)M * KVDIM * 2;
  bf16* Vt  = (bf16*)w; w += (size_t)M * KVDIM * 2;
  bf16* Ao  = (bf16*)w; w += (size_t)M * HID * 2;

  auto cvt = [&](const float* s, bf16* d, int n) {
    f32_to_bf16_kernel<<<n / 1024, 256, 0, stream>>>(s, d, n);
  };
  cvt(x,  Xb,  M * HID);
  cvt(Wq, Wqb, HID * HID);
  cvt(Wk, Wkb, KVDIM * HID);
  cvt(Wv, Wvb, KVDIM * HID);
  cvt(Wo, Wob, HID * HID);

  dim3 blk(256);
  // Q/K/V projections
  gemm_bf16_wmma<0><<<dim3(M / 64, HID / 256),   blk, 0, stream>>>(Xb, Wqb, bq, Qb, HID,   HID);
  gemm_bf16_wmma<0><<<dim3(M / 64, KVDIM / 256), blk, 0, stream>>>(Xb, Wkb, bk, Kb, KVDIM, HID);
  gemm_bf16_wmma<1><<<dim3(M / 64, KVDIM / 256), blk, 0, stream>>>(Xb, Wvb, bv, Vt, KVDIM, HID);
  // attention
  gqa_flash_attn<<<dim3(SEQ / 64, NHEADS, 2), dim3(128), 0, stream>>>(Qb, Kb, Vt, Ao);
  // output projection (f32 out)
  gemm_bf16_wmma<2><<<dim3(M / 64, HID / 256), blk, 0, stream>>>(Ao, Wob, bo, out, HID, HID);
}
